// TextualContrastiveLoss_29738353557639
// MI455X (gfx1250) — matrix operand order repaired and checked by reading, not compiled, not run on hardware
//
#include <hip/hip_runtime.h>
#include <hip/hip_bf16.h>

typedef __attribute__((ext_vector_type(16))) _Float16 v16h;
typedef __attribute__((ext_vector_type(8)))  _Float16 v8h;
typedef __attribute__((ext_vector_type(4)))  _Float16 v4h;
typedef __attribute__((ext_vector_type(8)))  float    v8f;

#define B_ROWS     4096
#define DIM        1024
#define N2         8192   // 2*B
#define JBLOCKS    64     // N2 / 128
#define INV_T      2.0f   // 1 / TEMPERATURE
#define LDS_STRIDE 40     // halves per column (32 data + 8 pad) -> conflict-free ds_load_b128
#define PANEL      (128 * LDS_STRIDE)

// ---------------------------------------------------------------------------
// Kernel 1: L2-normalize each row of [emb_i; emb_j] and emit f16 reps [8192,1024]
// ---------------------------------------------------------------------------
__global__ __launch_bounds__(256) void normalize_rows_kernel(
    const float* __restrict__ emb_i, const float* __restrict__ emb_j,
    _Float16* __restrict__ reps)
{
  const int row = blockIdx.x;
  const float* src = (row < B_ROWS) ? (emb_i + (size_t)row * DIM)
                                    : (emb_j + (size_t)(row - B_ROWS) * DIM);
  const int t = threadIdx.x;
  float4 x = ((const float4*)src)[t];
  float ss = x.x * x.x + x.y * x.y + x.z * x.z + x.w * x.w;
  #pragma unroll
  for (int off = 16; off >= 1; off >>= 1) ss += __shfl_xor(ss, off, 32);

  __shared__ float wsum[8];
  __shared__ float stotal;
  if ((t & 31) == 0) wsum[t >> 5] = ss;
  __syncthreads();
  if (t == 0) {
    float tot = 0.f;
    #pragma unroll
    for (int w = 0; w < 8; ++w) tot += wsum[w];
    stotal = tot;
  }
  __syncthreads();

  const float rn = 1.0f / fmaxf(sqrtf(stotal), 1e-12f);  // x / max(||x||, eps)
  v4h o;
  o[0] = (_Float16)(x.x * rn);
  o[1] = (_Float16)(x.y * rn);
  o[2] = (_Float16)(x.z * rn);
  o[3] = (_Float16)(x.w * rn);
  ((v4h*)(reps + (size_t)row * DIM))[t] = o;
}

// ---------------------------------------------------------------------------
// Kernel 2: fused Gram-matrix + masked row sums of exp(sim/T).
// Grid (64, 64): blockIdx.x = j-macro-tile, blockIdx.y = i-macro-tile (128x128).
// 256 threads = 8 waves; wave w owns rows [iBase+16w, iBase+16w+16) x 128 cols.
// Double-buffered, bank-conflict-free LDS B-panel; 8 back-to-back WMMAs per
// k-step with next-panel global loads in flight; one barrier per iteration.
// ---------------------------------------------------------------------------
__global__ __launch_bounds__(256) void sim_rowsum_kernel(
    const _Float16* __restrict__ reps, const int* __restrict__ labels,
    float* __restrict__ nomP, float* __restrict__ denP)
{
  __shared__ alignas(64) _Float16 ldsB[2 * PANEL];  // 2 x 10 KB B panels
  __shared__ int ldsLab[128];

  const int jBase   = blockIdx.x * 128;
  const int iBase   = blockIdx.y * 128;
  const int t       = threadIdx.x;
  const int wave    = t >> 5;
  const int lane    = t & 31;
  const int laneLo  = lane & 15;
  const int laneHi  = lane >> 4;
  const int rowBase = iBase + wave * 16;

  if (t < 128) ldsLab[t] = labels[(jBase + t) & (B_ROWS - 1)];

  // Labels for this lane's 8 output rows (C layout: row = r + laneHi*8).
  int labI[8];
  #pragma unroll
  for (int r = 0; r < 8; ++r)
    labI[r] = labels[(rowBase + r + laneHi * 8) & (B_ROWS - 1)];

  v8f zero = {};
  v8f acc[8];
  #pragma unroll
  for (int tl = 0; tl < 8; ++tl) acc[tl] = zero;

  // Cooperative fill mapping: 2 threads per column, 16 halves (32 B) each.
  const int fillCol = t >> 1;
  const int fillOfs = (t & 1) * 16;
  const _Float16* fillSrc = reps + (size_t)(jBase + fillCol) * DIM + fillOfs;
  const int fillLds = fillCol * LDS_STRIDE + fillOfs;

  // Prologue: fill panel 0 with k0 = 0.
  {
    const uint4* g = (const uint4*)(fillSrc);
    uint4 s0 = g[0], s1 = g[1];
    uint4* dvec = (uint4*)(ldsB + fillLds);
    dvec[0] = s0; dvec[1] = s1;
  }
  __syncthreads();

  for (int k0 = 0; k0 < DIM; k0 += 32) {
    const int cb = (k0 >> 5) & 1;
    const _Float16* cur = ldsB + cb * PANEL;

    // Stage next panel's global loads (issued early, consumed after WMMA chain).
    uint4 s0, s1;
    const bool haveNext = (k0 + 32) < DIM;
    if (haveNext) {
      const uint4* g = (const uint4*)(fillSrc + k0 + 32);
      s0 = g[0]; s1 = g[1];
    }

    // A fragment (16x32 f16): lane laneLo = row; lanes 0-15 hold K {0..7,16..23},
    // lanes 16-31 hold K {8..15,24..31} -> two contiguous 16-byte chunks.
    const _Float16* aBase = reps + (size_t)(rowBase + laneLo) * DIM + k0 + laneHi * 8;
    v8h a0 = *(const v8h*)(aBase);
    v8h a1 = *(const v8h*)(aBase + 16);
    v16h a;
    #pragma unroll
    for (int i = 0; i < 8; ++i) { a[i] = a0[i]; a[i + 8] = a1[i]; }

    // Preload ALL 8 B fragments into registers (conflict-free ds_load_b128),
    // so the 8 WMMAs below issue back-to-back with a single dscnt wait.
    v16h b[8];
    #pragma unroll
    for (int tl = 0; tl < 8; ++tl) {
      const _Float16* bp = cur + (tl * 16 + laneLo) * LDS_STRIDE + laneHi * 8;
      v8h b0 = *(const v8h*)(bp);
      v8h b1 = *(const v8h*)(bp + 16);
      #pragma unroll
      for (int i = 0; i < 8; ++i) { b[tl][i] = b0[i]; b[tl][i + 8] = b1[i]; }
    }

    #pragma unroll
    for (int tl = 0; tl < 8; ++tl) {
      acc[tl] = __builtin_amdgcn_wmma_f32_16x16x32_f16(
          /*neg_a=*/false, a, /*neg_b=*/false, b[tl],
          /*c_mod=*/(short)0, acc[tl], /*reuse_a=*/false, /*reuse_b=*/false);
    }

    // Write staged data into the other panel; single barrier per iteration.
    if (haveNext) {
      uint4* dvec = (uint4*)(ldsB + (cb ^ 1) * PANEL + fillLds);
      dvec[0] = s0; dvec[1] = s1;
    }
    __syncthreads();
  }

  // Epilogue: e = exp(sim/T); mask diagonal; label-match for nominator.
  float nomAcc[8], denAcc[8];
  #pragma unroll
  for (int r = 0; r < 8; ++r) { nomAcc[r] = 0.f; denAcc[r] = 0.f; }

  #pragma unroll
  for (int tl = 0; tl < 8; ++tl) {
    const int gj   = jBase + tl * 16 + laneLo;
    const int labJ = ldsLab[tl * 16 + laneLo];
    #pragma unroll
    for (int r = 0; r < 8; ++r) {
      const int gi = rowBase + r + laneHi * 8;
      float e = __expf(acc[tl][r] * INV_T);
      e = (gi != gj) ? e : 0.0f;                 // exclude self-pairs
      denAcc[r] += e;
      nomAcc[r] += (labI[r] == labJ) ? e : 0.0f; // same-label positives
    }
  }

  // Reduce over the 16 columns held by each half-wave (C layout: col = lane&15).
  #pragma unroll
  for (int off = 8; off >= 1; off >>= 1) {
    #pragma unroll
    for (int r = 0; r < 8; ++r) {
      denAcc[r] += __shfl_xor(denAcc[r], off, 32);
      nomAcc[r] += __shfl_xor(nomAcc[r], off, 32);
    }
  }

  // Lane 0 holds rows rowBase..+7, lane 16 rows rowBase+8..+15.
  // Exactly one lane writes each (jblock,row) -> deterministic, no atomics.
  if (laneLo == 0) {
    #pragma unroll
    for (int r = 0; r < 8; ++r) {
      const int gi = rowBase + r + laneHi * 8;
      nomP[(size_t)blockIdx.x * N2 + gi] = nomAcc[r];
      denP[(size_t)blockIdx.x * N2 + gi] = denAcc[r];
    }
  }
}

// ---------------------------------------------------------------------------
// Kernel 3: loss = mean_i( -log(nom_i / den_i) ), single block, deterministic.
// ---------------------------------------------------------------------------
__global__ __launch_bounds__(256) void finalize_kernel(
    const float* __restrict__ nomP, const float* __restrict__ denP,
    float* __restrict__ out)
{
  __shared__ float red[256];
  const int t = threadIdx.x;
  float local = 0.f;
  for (int i = t; i < N2; i += 256) {
    float n = 0.f, d = 0.f;
    #pragma unroll 8
    for (int jb = 0; jb < JBLOCKS; ++jb) {
      n += nomP[(size_t)jb * N2 + i];
      d += denP[(size_t)jb * N2 + i];
    }
    local += -__logf(n / d);
  }
  red[t] = local;
  __syncthreads();
  for (int s = 128; s >= 1; s >>= 1) {
    if (t < s) red[t] += red[t + s];
    __syncthreads();
  }
  if (t == 0) out[0] = red[0] * (1.0f / (float)N2);
}

// ---------------------------------------------------------------------------
extern "C" void kernel_launch(void* const* d_in, const int* in_sizes, int n_in,
                              void* d_out, int out_size, void* d_ws, size_t ws_size,
                              hipStream_t stream) {
  (void)in_sizes; (void)n_in; (void)out_size; (void)ws_size;
  const float* emb_i  = (const float*)d_in[0];
  const float* emb_j  = (const float*)d_in[1];
  const int*   labels = (const int*)d_in[2];
  float* out = (float*)d_out;

  // Workspace layout:
  //   [0, 16MB)            : f16 reps [8192 x 1024]
  //   [16MB, 18MB)         : nomP [64 x 8192] f32
  //   [18MB, 20MB)         : denP [64 x 8192] f32
  char* ws = (char*)d_ws;
  _Float16* reps = (_Float16*)ws;
  float* nomP = (float*)(ws + (size_t)N2 * DIM * sizeof(_Float16));
  float* denP = nomP + (size_t)JBLOCKS * N2;

  normalize_rows_kernel<<<N2, 256, 0, stream>>>(emb_i, emb_j, reps);
  sim_rowsum_kernel<<<dim3(JBLOCKS, JBLOCKS), 256, 0, stream>>>(reps, labels, nomP, denP);
  finalize_kernel<<<1, 256, 0, stream>>>(nomP, denP, out);
}
